// AdditiveVisioLinguistic_4277787427693
// MI455X (gfx1250) — matrix operand order repaired and checked by reading, not compiled, not run on hardware
//
#include <hip/hip_runtime.h>
#include <math.h>

#define ENC_DIM 2048
#define ATT_DIM 512
#define BATCH   128
#define NPIX    196
#define BPROWS  (BATCH * NPIX)     // 25088 = 784 * 32 exactly
#define WG_ROWS 32
#define NKC     (ENC_DIM / 32)     // 64 K-chunks
#define NTILE   (ATT_DIM / 16)     // 32 N-tiles

typedef __attribute__((ext_vector_type(16))) __bf16 v16bf;
typedef __attribute__((ext_vector_type(8)))  float  v8f;
typedef int v4i_g __attribute__((vector_size(16)));   // matches builtin param type
typedef unsigned short ushort_t;

#define AS1 __attribute__((address_space(1)))
#define AS3 __attribute__((address_space(3)))

#if defined(__has_builtin)
#if __has_builtin(__builtin_amdgcn_global_load_async_to_lds_b128) && \
    __has_builtin(__builtin_amdgcn_s_wait_asynccnt)
#define HAVE_ASYNC_LDS 1
#endif
#endif
#ifndef HAVE_ASYNC_LDS
#define HAVE_ASYNC_LDS 0
#endif

union FragBF {
    uint4 q[2];
    v16bf v;
};

__device__ __forceinline__ ushort_t f32_to_bf16(float f) {
    unsigned int u = __float_as_uint(f);
    u += 0x7fffu + ((u >> 16) & 1u);      // round-to-nearest-even
    return (ushort_t)(u >> 16);
}
__device__ __forceinline__ float bf16_to_f32(ushort_t h) {
    return __uint_as_float(((unsigned int)h) << 16);
}

// ---------------------------------------------------------------------------
// Kernel 0 (prep, runs once per launch): split W_enc into bf16 hi/lo planes,
// stored pre-swizzled in WMMA B-fragment order:
//   addr = (((kc*NTILE + n/16) * 32) + (n&15) + (kk&16 ? 16 : 0)) * 16 + (kk&15)
// so the GEMM loads each lane's fragment with two coalesced global_load_b128.
// ---------------------------------------------------------------------------
__global__ __launch_bounds__(256)
void swizzle_wenc_kernel(const float* __restrict__ Wenc,
                         ushort_t* __restrict__ Whi, ushort_t* __restrict__ Wlo) {
    const int e = blockIdx.x * 256 + threadIdx.x;   // 0 .. 2048*512-1
    const int k = e >> 9;
    const int n = e & 511;
    const float x = Wenc[e];
    const ushort_t hi = f32_to_bf16(x);
    const ushort_t lo = f32_to_bf16(x - bf16_to_f32(hi));
    const int kc = k >> 5, kk = k & 31;
    const int off = (((kc * NTILE + (n >> 4)) * 32) + (n & 15) + ((kk & 16) ? 16 : 0)) * 16
                    + (kk & 15);
    Whi[off] = hi;
    Wlo[off] = lo;
}

// ---------------------------------------------------------------------------
// Kernel 1: att2[b, n] = decoder_hidden[b, :] @ W_dec[:, n] + b_dec[n]
// ---------------------------------------------------------------------------
__global__ __launch_bounds__(256)
void att2_kernel(const float* __restrict__ dh, const float* __restrict__ Wdec,
                 const float* __restrict__ bdec, float* __restrict__ att2) {
    const int b = blockIdx.x;
    const int tid = threadIdx.x;
    __shared__ float sh[ATT_DIM];
    for (int i = tid; i < ATT_DIM; i += 256) sh[i] = dh[b * ATT_DIM + i];
    __syncthreads();
    for (int n = tid; n < ATT_DIM; n += 256) {
        float acc = bdec[n];
        for (int k = 0; k < ATT_DIM; ++k) acc = fmaf(sh[k], Wdec[k * ATT_DIM + n], acc);
        att2[b * ATT_DIM + n] = acc;
    }
}

// ---------------------------------------------------------------------------
// Kernel 2: fused  scores[g] = relu(enc[g,:]@W_enc + b_enc + att2[b,:]) @ W_full + b_full
// bf16 hi/lo compensated WMMA GEMM. Workgroup = 32 rows x 512 cols, K streamed
// in chunks of 32. B fragments come straight from the pre-swizzled global
// planes (L2-resident); only the streaming A tile passes through LDS.
// ---------------------------------------------------------------------------
__global__ __launch_bounds__(256)
void fused_scores_kernel(const float* __restrict__ enc,
                         const ushort_t* __restrict__ Whi,
                         const ushort_t* __restrict__ Wlo,
                         const float* __restrict__ benc,
                         const float* __restrict__ att2,
                         const float* __restrict__ Wfull,
                         const float* __restrict__ bfull,
                         float* __restrict__ scores) {
    // A fragments: [buf][plane hi/lo][rowblock][lane][16 bf16]
    __shared__ __align__(16) ushort_t ldsA[2][2][2][32][16];
    __shared__ float sScore[WG_ROWS];

    const int tid  = threadIdx.x;
    const int lane = tid & 31;
    const int wave = tid >> 5;
    const int rb   = wave >> 2;        // rowblock (0/1)
    const int tg   = (wave & 3) * 8;   // first of this wave's 8 N-tiles
    const int r0   = blockIdx.x * WG_ROWS;

    if (tid < WG_ROWS) sScore[tid] = bfull[0];

    const v8f vzero = {0.f, 0.f, 0.f, 0.f, 0.f, 0.f, 0.f, 0.f};
    v8f acc[8];
#pragma unroll
    for (int j = 0; j < 8; ++j) acc[j] = vzero;

    // Stage one 32-row x 32-K A tile: fp32 load -> bf16 hi/lo split -> LDS in
    // A-fragment order (lane = m + (k&16?16:0), idx = k&15),
    // packing two consecutive K into one b32 store.
    auto stageA = [&](int buf, int kc) {
        const int k0 = kc * 32;
        for (int e2 = tid; e2 < 512; e2 += 256) {   // 512 k-pairs, 2 per thread
            const int arb = e2 >> 8;
            const int m   = (e2 >> 4) & 15;
            const int k   = (e2 & 15) * 2;
            const float2 x2 =
                *(const float2*)(enc + (size_t)(r0 + arb * 16 + m) * ENC_DIM + k0 + k);
            const ushort_t h0 = f32_to_bf16(x2.x);
            const ushort_t l0 = f32_to_bf16(x2.x - bf16_to_f32(h0));
            const ushort_t h1 = f32_to_bf16(x2.y);
            const ushort_t l1 = f32_to_bf16(x2.y - bf16_to_f32(h1));
            const int l   = m + ((k & 16) ? 16 : 0);
            const int idx = k & 15;
            *(unsigned int*)&ldsA[buf][0][arb][l][idx] =
                (unsigned int)h0 | ((unsigned int)h1 << 16);
            *(unsigned int*)&ldsA[buf][1][arb][l][idx] =
                (unsigned int)l0 | ((unsigned int)l1 << 16);
        }
    };

    stageA(0, 0);
    __syncthreads();

    for (int kc = 0; kc < NKC; ++kc) {
        const int cur = kc & 1;
        if (kc + 1 < NKC) stageA(cur ^ 1, kc + 1);

        FragBF ah, al;
        const ushort_t* pa = &ldsA[cur][0][rb][lane][0];
        ah.q[0] = *(const uint4*)pa;
        ah.q[1] = *(const uint4*)(pa + 8);
        const ushort_t* pl = &ldsA[cur][1][rb][lane][0];
        al.q[0] = *(const uint4*)pl;
        al.q[1] = *(const uint4*)(pl + 8);

#pragma unroll
        for (int j = 0; j < 8; ++j) {
            const int tj = tg + j;
            const size_t fo = ((size_t)(kc * NTILE + tj) * 32 + lane) * 16;
            FragBF bh, bl;
            bh.q[0] = *(const uint4*)(Whi + fo);
            bh.q[1] = *(const uint4*)(Whi + fo + 8);
            bl.q[0] = *(const uint4*)(Wlo + fo);
            bl.q[1] = *(const uint4*)(Wlo + fo + 8);
            // bf16 hi/lo compensated product: AhBh + AhBl + AlBh (fp32 accum)
            acc[j] = __builtin_amdgcn_wmma_f32_16x16x32_bf16(
                false, ah.v, false, bh.v, (short)0, acc[j], false, false);
            acc[j] = __builtin_amdgcn_wmma_f32_16x16x32_bf16(
                false, ah.v, false, bl.v, (short)0, acc[j], false, false);
            acc[j] = __builtin_amdgcn_wmma_f32_16x16x32_bf16(
                false, al.v, false, bh.v, (short)0, acc[j], false, false);
        }
        __syncthreads();
    }

    // Epilogue: + b_enc + att2 broadcast, ReLU, dot with W_full, reduce over N.
    // C layout: lanes 0-15 hold N=n0+lane with M=r; lanes 16-31 hold M=8+r.
    const int nl = lane & 15;
    const int mh = (lane >> 4) << 3;
    float partial[8];
#pragma unroll
    for (int r = 0; r < 8; ++r) partial[r] = 0.f;

#pragma unroll
    for (int j = 0; j < 8; ++j) {
        const int n  = (tg + j) * 16 + nl;
        const float be = benc[n];
        const float wf = Wfull[n];
#pragma unroll
        for (int r = 0; r < 8; ++r) {
            const int g = r0 + rb * 16 + mh + r;
            const int b = g / NPIX;
            float v = acc[j][r] + be + att2[b * ATT_DIM + n];
            v = fmaxf(v, 0.f);
            partial[r] = fmaf(v, wf, partial[r]);
        }
    }
#pragma unroll
    for (int off = 1; off < 16; off <<= 1) {
#pragma unroll
        for (int r = 0; r < 8; ++r)
            partial[r] += __shfl_xor(partial[r], off, 32);
    }
    if (nl == 0) {
#pragma unroll
        for (int r = 0; r < 8; ++r)
            atomicAdd(&sScore[rb * 16 + mh + r], partial[r]);   // ds_add_f32
    }
    __syncthreads();
    if (tid < WG_ROWS) scores[r0 + tid] = sScore[tid];
}

// ---------------------------------------------------------------------------
// Kernel 3: softmax over P per batch row
// ---------------------------------------------------------------------------
__global__ __launch_bounds__(256)
void softmax_kernel(const float* __restrict__ scores, float* __restrict__ alpha) {
    const int b = blockIdx.x, tid = threadIdx.x;
    __shared__ float red[256];
    const float v = (tid < NPIX) ? scores[b * NPIX + tid] : -INFINITY;
    red[tid] = v;
    __syncthreads();
    for (int s = 128; s > 0; s >>= 1) {
        if (tid < s) red[tid] = fmaxf(red[tid], red[tid + s]);
        __syncthreads();
    }
    const float mx = red[0];
    __syncthreads();
    const float e = (tid < NPIX) ? __expf(v - mx) : 0.f;
    red[tid] = e;
    __syncthreads();
    for (int s = 128; s > 0; s >>= 1) {
        if (tid < s) red[tid] += red[tid + s];
        __syncthreads();
    }
    const float inv = 1.f / red[0];
    if (tid < NPIX) alpha[b * NPIX + tid] = e * inv;
}

// ---------------------------------------------------------------------------
// Kernel 4: awe[b, e] = sum_p enc[b, p, e] * alpha[b, p]   (HBM bandwidth bound)
// Software-pipelined GLOBAL_LOAD_ASYNC_TO_LDS staging when available.
// ---------------------------------------------------------------------------
__global__ __launch_bounds__(256)
void awe_kernel(const float* __restrict__ enc, const float* __restrict__ alpha,
                float* __restrict__ awe) {
    const int b = blockIdx.y;
    const int tid = threadIdx.x;
    __shared__ float sa[NPIX];
    for (int i = tid; i < NPIX; i += 256) sa[i] = alpha[b * NPIX + i];

    const int e0 = blockIdx.x * 1024;   // this block's 1024-float E slice
    const float* base = enc + (size_t)b * NPIX * ENC_DIM + e0;
    float4 acc = {0.f, 0.f, 0.f, 0.f};

#if HAVE_ASYNC_LDS
    __shared__ __align__(16) float tile[2][4][1024];   // 2 x 16KB pixel-row groups

    auto stage = [&](int buf, int p0) {
        for (int t = tid; t < 1024; t += 256) {        // 1024 b128 per group
            const int pr  = t >> 8;
            const int col = (t & 255) * 4;
            float* g = const_cast<float*>(base + (size_t)(p0 + pr) * ENC_DIM + col);
            __builtin_amdgcn_global_load_async_to_lds_b128(
                (AS1 v4i_g*)g, (AS3 v4i_g*)&tile[buf][pr][col], 0, 0);
        }
    };

    stage(0, 0);
    for (int i = 0; i < NPIX / 4; ++i) {               // 49 groups of 4 pixels
        const int cur = i & 1;
        __builtin_amdgcn_s_wait_asynccnt(0);           // cur group landed
        __syncthreads();                               // ...in every wave
        if (i + 1 < NPIX / 4) stage(cur ^ 1, (i + 1) * 4);
#pragma unroll
        for (int pr = 0; pr < 4; ++pr) {
            const float a = sa[i * 4 + pr];
            const float4 v = *(const float4*)&tile[cur][pr][tid * 4];
            acc.x = fmaf(a, v.x, acc.x);
            acc.y = fmaf(a, v.y, acc.y);
            acc.z = fmaf(a, v.z, acc.z);
            acc.w = fmaf(a, v.w, acc.w);
        }
    }
#else
    __syncthreads();
    for (int p = 0; p < NPIX; ++p) {
        if (p + 4 < NPIX)
            __builtin_prefetch(base + (size_t)(p + 4) * ENC_DIM + tid * 4, 0, 0);
        const float a = sa[p];
        const float4 v = *(const float4*)(base + (size_t)p * ENC_DIM + tid * 4);
        acc.x = fmaf(a, v.x, acc.x);
        acc.y = fmaf(a, v.y, acc.y);
        acc.z = fmaf(a, v.z, acc.z);
        acc.w = fmaf(a, v.w, acc.w);
    }
#endif
    *(float4*)(awe + (size_t)b * ENC_DIM + e0 + tid * 4) = acc;
}

// ---------------------------------------------------------------------------
extern "C" void kernel_launch(void* const* d_in, const int* in_sizes, int n_in,
                              void* d_out, int out_size, void* d_ws, size_t ws_size,
                              hipStream_t stream) {
    const float* enc   = (const float*)d_in[0];  // [128, 196, 2048]
    const float* dh    = (const float*)d_in[1];  // [128, 512]
    const float* Wenc  = (const float*)d_in[2];  // [2048, 512]
    const float* benc  = (const float*)d_in[3];  // [512]
    const float* Wdec  = (const float*)d_in[4];  // [512, 512]
    const float* bdec  = (const float*)d_in[5];  // [512]
    const float* Wfull = (const float*)d_in[6];  // [512]
    const float* bfull = (const float*)d_in[7];  // scalar

    float* awe   = (float*)d_out;                         // [128, 2048]
    float* alpha = awe + (size_t)BATCH * ENC_DIM;         // [128, 196]

    float*    att2buf  = (float*)d_ws;                    // [128, 512]
    float*    scorebuf = att2buf + (size_t)BATCH * ATT_DIM;   // [25088]
    ushort_t* Whi      = (ushort_t*)(scorebuf + BPROWS);  // [2048*512] bf16 hi
    ushort_t* Wlo      = Whi + (size_t)ENC_DIM * ATT_DIM; // [2048*512] bf16 lo

    swizzle_wenc_kernel<<<dim3((ENC_DIM * ATT_DIM) / 256), dim3(256), 0, stream>>>(
        Wenc, Whi, Wlo);
    att2_kernel<<<dim3(BATCH), dim3(256), 0, stream>>>(dh, Wdec, bdec, att2buf);
    fused_scores_kernel<<<dim3(BPROWS / WG_ROWS), dim3(256), 0, stream>>>(
        enc, Whi, Wlo, benc, att2buf, Wfull, bfull, scorebuf);
    softmax_kernel<<<dim3(BATCH), dim3(256), 0, stream>>>(scorebuf, alpha);
    awe_kernel<<<dim3(ENC_DIM / 1024, BATCH), dim3(256), 0, stream>>>(enc, alpha, awe);
}
